// Graph2SeqDecoder_67577015435340
// MI455X (gfx1250) — matrix operand order repaired and checked by reading, not compiled, not run on hardware
//
#include <hip/hip_runtime.h>
#include <hip/hip_bf16.h>

typedef __attribute__((ext_vector_type(16))) __bf16 v16bf;
typedef __attribute__((ext_vector_type(8)))  float  v8f;

#define CDIM 64
#define TQ   2048
#define NN   1024

// Branch-free gelu: Abramowitz-Stegun 7.1.26 erf approximation (|err| < 1.5e-7).
// One v_rcp, one v_exp, 5 FMA -- no divergent libm slow paths, tiny live range.
__device__ __forceinline__ float gelu_f(float x) {
    float a = fabsf(x);
    float t = __builtin_amdgcn_rcpf(__builtin_fmaf(0.3275911f, a, 1.0f));
    float p = t * (0.254829592f +
              t * (-0.284496736f +
              t * (1.421413741f +
              t * (-1.453152027f +
              t * 1.061405429f))));
    float e = __expf(-a * a);
    float erfv = copysignf(__builtin_fmaf(-p, e, 1.0f), x);
    return 0.5f * x * (1.0f + erfv);
}

__device__ __forceinline__ __bf16 f2bf(float f) {
    union { float f; unsigned u; } v; v.f = f;
    unsigned r = v.u + 0x7FFFu + ((v.u >> 16) & 1u);
    unsigned short h = (unsigned short)(r >> 16);
    __bf16 o;
    __builtin_memcpy(&o, &h, 2);
    return o;
}

// ---------------------------------------------------------------------------
// Kernel 1: query embedding. z0[t, :] = gelu(inp @ W1.T + b1) @ W2.T + b2
// inp = [qpts(3), sin/cos fourier(48)].   Grid: TQ blocks x 64 threads.
// ---------------------------------------------------------------------------
__global__ void qe_kernel(const float* __restrict__ qpts,
                          const float* __restrict__ W1, const float* __restrict__ b1,
                          const float* __restrict__ W2, const float* __restrict__ b2,
                          float* __restrict__ z0) {
    const int t = blockIdx.x, tid = threadIdx.x;
    __shared__ float inp[51];
    __shared__ float h1[64];
    if (tid < 3) inp[tid] = qpts[t * 3 + tid];
    if (tid < 48) {
        int d = tid / 16, j = tid % 16;
        float f = (float)(1 << (j & 7)) * 3.14159265358979323846f;
        float v = qpts[t * 3 + d] * f;
        inp[3 + tid] = (j < 8) ? sinf(v) : cosf(v);
    }
    __syncthreads();
    {
        float s = b1[tid];
        #pragma unroll 17
        for (int k = 0; k < 51; ++k) s += inp[k] * W1[tid * 51 + k];
        h1[tid] = gelu_f(s);
    }
    __syncthreads();
    {
        float s = b2[tid];
        #pragma unroll 16
        for (int k = 0; k < 64; ++k) s += h1[k] * W2[tid * 64 + k];
        z0[t * 64 + tid] = s;
    }
}

// ---------------------------------------------------------------------------
// Kernel 2: K = H @ Wk.T, V = H @ Wv.T.   Grid: NN blocks x 64 threads.
// ---------------------------------------------------------------------------
__global__ void kv_kernel(const float* __restrict__ H,
                          const float* __restrict__ Wk, const float* __restrict__ Wv,
                          float* __restrict__ kb, float* __restrict__ vb) {
    const int n = blockIdx.x, tid = threadIdx.x;
    __shared__ float hrow[64];
    hrow[tid] = H[n * 64 + tid];
    __syncthreads();
    float kk = 0.f, vv = 0.f;
    #pragma unroll 16
    for (int c = 0; c < 64; ++c) {
        kk += hrow[c] * Wk[tid * 64 + c];
        vv += hrow[c] * Wv[tid * 64 + c];
    }
    kb[n * 64 + tid] = kk;
    vb[n * 64 + tid] = vv;
}

// ---------------------------------------------------------------------------
// Kernel 3: fused attention layer (q proj, 4-head attention over N=1024 with
// online softmax, out proj, +res, LN1, FFN(256), +res, LN2).
// Grid: TQ blocks x 128 threads (one wave per head).
// ---------------------------------------------------------------------------
__global__ void attn_layer_kernel(const float* __restrict__ zin, float* __restrict__ zout,
                                  const float* __restrict__ kb, const float* __restrict__ vb,
                                  const float* __restrict__ Wq, const float* __restrict__ Wout,
                                  const float* __restrict__ ln1g, const float* __restrict__ ln1b,
                                  const float* __restrict__ fW1, const float* __restrict__ fb1,
                                  const float* __restrict__ fW2, const float* __restrict__ fb2,
                                  const float* __restrict__ ln2g, const float* __restrict__ ln2b) {
    const int t = blockIdx.x, tid = threadIdx.x;
    const int w = tid >> 5, lane = tid & 31;
    __shared__ float zrow[64], qv[64], ov[64], xrow[64], buf[64], hsh[256], red[2];

    if (tid < 64) zrow[tid] = zin[t * 64 + tid];
    __syncthreads();
    if (tid < 64) {
        float s = 0.f;
        #pragma unroll 16
        for (int c = 0; c < 64; ++c) s += zrow[c] * Wq[tid * 64 + c];
        qv[tid] = s;
    }
    __syncthreads();

    // per-head attention: wave w == head w
    {
        const int h = w;
        float q[16];
        #pragma unroll
        for (int d = 0; d < 16; ++d) q[d] = qv[h * 16 + d];
        float m = -1e30f, l = 0.f, acc[16];
        #pragma unroll
        for (int d = 0; d < 16; ++d) acc[d] = 0.f;
        for (int n = lane; n < NN; n += 32) {
            const float* kp = kb + n * 64 + h * 16;
            float s = 0.f;
            #pragma unroll
            for (int d = 0; d < 16; ++d) s += q[d] * kp[d];
            s *= 0.25f;  // 1/sqrt(16)
            float nm = fmaxf(m, s);
            float corr = __expf(m - nm);
            float p = __expf(s - nm);
            l = l * corr + p;
            const float* vp = vb + n * 64 + h * 16;
            #pragma unroll
            for (int d = 0; d < 16; ++d) acc[d] = acc[d] * corr + p * vp[d];
            m = nm;
        }
        float gm = m;
        #pragma unroll
        for (int mk = 16; mk >= 1; mk >>= 1) gm = fmaxf(gm, __shfl_xor(gm, mk));
        float c = __expf(m - gm);
        l *= c;
        #pragma unroll
        for (int d = 0; d < 16; ++d) acc[d] *= c;
        #pragma unroll
        for (int mk = 16; mk >= 1; mk >>= 1) {
            l += __shfl_xor(l, mk);
            #pragma unroll
            for (int d = 0; d < 16; ++d) acc[d] += __shfl_xor(acc[d], mk);
        }
        if (lane == 0) {
            float inv = 1.f / l;
            #pragma unroll
            for (int d = 0; d < 16; ++d) ov[h * 16 + d] = acc[d] * inv;
        }
    }
    __syncthreads();

    // out proj + residual
    if (tid < 64) {
        float s = 0.f;
        #pragma unroll 16
        for (int c = 0; c < 64; ++c) s += ov[c] * Wout[tid * 64 + c];
        buf[tid] = s + zrow[tid];
    }
    __syncthreads();
    if (tid == 0) {
        float sm = 0.f, sq = 0.f;
        for (int c = 0; c < 64; ++c) { float v = buf[c]; sm += v; sq += v * v; }
        float mean = sm * (1.f / 64.f);
        red[0] = mean;
        red[1] = rsqrtf(sq * (1.f / 64.f) - mean * mean + 1e-5f);
    }
    __syncthreads();
    if (tid < 64) xrow[tid] = (buf[tid] - red[0]) * red[1] * ln1g[tid] + ln1b[tid];
    __syncthreads();

    // FFN
    for (int u = tid; u < 256; u += 128) {
        float s = fb1[u];
        #pragma unroll 16
        for (int c = 0; c < 64; ++c) s += xrow[c] * fW1[u * 64 + c];
        hsh[u] = gelu_f(s);
    }
    __syncthreads();
    if (tid < 64) {
        float s = fb2[tid];
        #pragma unroll 16
        for (int u = 0; u < 256; ++u) s += hsh[u] * fW2[tid * 256 + u];
        buf[tid] = s + xrow[tid];
    }
    __syncthreads();
    if (tid == 0) {
        float sm = 0.f, sq = 0.f;
        for (int c = 0; c < 64; ++c) { float v = buf[c]; sm += v; sq += v * v; }
        float mean = sm * (1.f / 64.f);
        red[0] = mean;
        red[1] = rsqrtf(sq * (1.f / 64.f) - mean * mean + 1e-5f);
    }
    __syncthreads();
    if (tid < 64) zout[t * 64 + tid] = (buf[tid] - red[0]) * red[1] * ln2g[tid] + ln2b[tid];
}

// ---------------------------------------------------------------------------
// Kernel 4: pairwise prep. az = z @ pm_W1[:, :64].T, aHb = H @ pm_W1[:,64:].T + b1,
// W2' = bf16(pm_W2 * ln_g) (LN gain folded), b2' = pm_b2 + ln_b @ pm_W2.T,
// csum[nout] = sum_k W2'[nout,k] (for post-WMMA mean subtraction).
// Grid: (TQ + NN + 1) blocks x 64 threads.
// ---------------------------------------------------------------------------
__global__ void pm_prep_kernel(const float* __restrict__ z, const float* __restrict__ H,
                               const float* __restrict__ pmW1, const float* __restrict__ pmb1,
                               const float* __restrict__ lng, const float* __restrict__ lnb,
                               const float* __restrict__ pmW2, const float* __restrict__ pmb2,
                               float* __restrict__ az, float* __restrict__ aHb,
                               __bf16* __restrict__ W2bf, float* __restrict__ b2f,
                               float* __restrict__ csumf) {
    const int b = blockIdx.x, tid = threadIdx.x;
    __shared__ float row[64];
    if (b < TQ) {
        row[tid] = z[b * 64 + tid];
        __syncthreads();
        float s = 0.f;
        #pragma unroll 16
        for (int k = 0; k < 64; ++k) s += row[k] * pmW1[tid * 128 + k];
        az[b * 64 + tid] = s;
    } else if (b < TQ + NN) {
        const int n = b - TQ;
        row[tid] = H[n * 64 + tid];
        __syncthreads();
        float s = pmb1[tid];
        #pragma unroll 16
        for (int k = 0; k < 64; ++k) s += row[k] * pmW1[tid * 128 + 64 + k];
        aHb[n * 64 + tid] = s;
    } else {
        // thread = output row (nout)
        float bb = pmb2[tid];
        float cs = 0.f;
        #pragma unroll 16
        for (int k = 0; k < 64; ++k) {
            float wv = pmW2[tid * 64 + k];
            float wg = wv * lng[k];
            bb += lnb[k] * wv;
            cs += wg;
            W2bf[tid * 64 + k] = f2bf(wg);
        }
        b2f[tid] = bb;
        csumf[tid] = cs;
    }
}

// ---------------------------------------------------------------------------
// Kernel 5: pairwise head via WMMA bf16.
// h = gelu(az[t]+aHb[n]);  LN folded post-GEMM:
//   hn @ W2' = rstd * (h @ W2' - mean * csum)          (per output column)
// p2 = gelu(that + b2');  out[t,n,0:3] = p2 @ pm_W3.T + b3.
// Wave owns a 16-node tile (A rows), loops over 16 t's. 8 v_wmma per t-iter.
// Grid: 1024 blocks (64 n-tiles x 16 t-groups) x 256 threads (8 waves).
// ---------------------------------------------------------------------------
__global__ void __launch_bounds__(256, 1)
pairwise_kernel(const float* __restrict__ az, const float* __restrict__ aHb,
                const __bf16* __restrict__ W2bf, const float* __restrict__ b2f,
                const float* __restrict__ csumf,
                const float* __restrict__ pmW3, const float* __restrict__ pmb3,
                float* __restrict__ out) {
    const int lane  = threadIdx.x & 31;
    const int w     = threadIdx.x >> 5;     // wave in block: 0..7
    const int ntile = blockIdx.x & 63;      // 64 node tiles
    const int tg    = blockIdx.x >> 6;      // 16 t-groups of 128
    const int n0    = ntile * 16;
    const int half  = lane >> 4;            // K-half selector (A layout)
    const int mlane = lane & 15;            // A row / C column within tile
    const int node  = n0 + mlane;

    // invariant: aHb values for this lane's 32 K indices (4 runs of 8)
    float ah[32];
    {
        const float* p = aHb + node * 64;
        #pragma unroll
        for (int g = 0; g < 4; ++g) {
            const int kb = 16 * half + ((g & 1) ? 8 : 0) + ((g & 2) ? 32 : 0);
            #pragma unroll
            for (int o = 0; o < 8; ++o) ah[g * 8 + o] = p[kb + o];
        }
    }
    // invariant: B fragments (4 column tiles x 2 K-slabs), each one contiguous
    // 32-byte aligned run of 16 bf16 -> vector load.
    v16bf Bf[4][2];
    #pragma unroll
    for (int ct = 0; ct < 4; ++ct) {
        const int nout = ct * 16 + mlane;
        #pragma unroll
        for (int s = 0; s < 2; ++s) {
            Bf[ct][s] = *reinterpret_cast<const v16bf*>(W2bf + nout * 64 + 32 * s + 16 * half);
        }
    }
    float b2v[4], csv[4], w3v[3][4];
    #pragma unroll
    for (int ct = 0; ct < 4; ++ct) {
        b2v[ct] = b2f[ct * 16 + mlane];
        csv[ct] = csumf[ct * 16 + mlane];
        #pragma unroll
        for (int c = 0; c < 3; ++c) w3v[c][ct] = pmW3[c * 64 + ct * 16 + mlane];
    }
    const float b30 = pmb3[0], b31 = pmb3[1], b32 = pmb3[2];

    for (int it = 0; it < 16; ++it) {
        const int t = tg * 128 + w * 16 + it;
        const float* azp = az + t * 64;
        if (it + 1 < 16) __builtin_prefetch(az + (t + 1) * 64, 0, 1);

        // h = gelu(az + aHb); build A fragments directly (raw h, bf16) while
        // accumulating LN stats over this lane's 32 K's.
        float sm = 0.f, sq = 0.f;
        v16bf a0, a1;
        #pragma unroll
        for (int g = 0; g < 4; ++g) {
            const int kb = 16 * half + ((g & 1) ? 8 : 0) + ((g & 2) ? 32 : 0);
            #pragma unroll
            for (int o = 0; o < 8; ++o) {
                float x = azp[kb + o] + ah[g * 8 + o];
                float gl = gelu_f(x);
                sm += gl;
                sq += gl * gl;
                const int q = g * 8 + o;
                if (q < 16) a0[q] = f2bf(gl);
                else        a1[q - 16] = f2bf(gl);
            }
        }
        // partner lane (lane^16) holds the complementary 32 K's of same row
        sm += __shfl_xor(sm, 16);
        sq += __shfl_xor(sq, 16);
        const float mean = sm * (1.f / 64.f);
        const float rstd = rsqrtf(sq * (1.f / 64.f) - mean * mean + 1e-5f);

        float bs[24];
        #pragma unroll
        for (int i = 0; i < 24; ++i) bs[i] = 0.f;

        #pragma unroll
        for (int ct = 0; ct < 4; ++ct) {
            v8f acc = {};
            acc = __builtin_amdgcn_wmma_f32_16x16x32_bf16(false, a0, false, Bf[ct][0],
                                                          (short)0, acc, false, false);
            acc = __builtin_amdgcn_wmma_f32_16x16x32_bf16(false, a1, false, Bf[ct][1],
                                                          (short)0, acc, false, false);
            const float cb = mean * csv[ct];
            #pragma unroll
            for (int r = 0; r < 8; ++r) {
                float val = (acc[r] - cb) * rstd + b2v[ct];
                float gl = gelu_f(val);
                bs[0 * 8 + r] += gl * w3v[0][ct];
                bs[1 * 8 + r] += gl * w3v[1][ct];
                bs[2 * 8 + r] += gl * w3v[2][ct];
            }
        }
        // reduce over the 16 lanes sharing each row half
        #pragma unroll
        for (int mk = 1; mk < 16; mk <<= 1) {
            #pragma unroll
            for (int i = 0; i < 24; ++i) bs[i] += __shfl_xor(bs[i], mk);
        }
        if (mlane == 0) {
            // rows half*8 .. half*8+7, 3 floats each -> 24 contiguous floats
            float* op = out + (size_t)t * 3072 + (size_t)(n0 + half * 8) * 3;
            #pragma unroll
            for (int r = 0; r < 8; ++r) {
                op[r * 3 + 0] = bs[0 * 8 + r] + b30;
                op[r * 3 + 1] = bs[1 * 8 + r] + b31;
                op[r * 3 + 2] = bs[2 * 8 + r] + b32;
            }
        }
    }
}

// ---------------------------------------------------------------------------
extern "C" void kernel_launch(void* const* d_in, const int* in_sizes, int n_in,
                              void* d_out, int out_size, void* d_ws, size_t ws_size,
                              hipStream_t stream) {
    (void)in_sizes; (void)n_in; (void)out_size; (void)ws_size;
    const float* H       = (const float*)d_in[0];
    const float* qpts    = (const float*)d_in[1];
    const float* qe_W1   = (const float*)d_in[2];
    const float* qe_b1   = (const float*)d_in[3];
    const float* qe_W2   = (const float*)d_in[4];
    const float* qe_b2   = (const float*)d_in[5];
    const float* attn_in = (const float*)d_in[6];
    const float* attn_out= (const float*)d_in[7];
    const float* ln1g    = (const float*)d_in[8];
    const float* ln1b    = (const float*)d_in[9];
    const float* fW1     = (const float*)d_in[10];
    const float* fb1     = (const float*)d_in[11];
    const float* fW2     = (const float*)d_in[12];
    const float* fb2     = (const float*)d_in[13];
    const float* ln2g    = (const float*)d_in[14];
    const float* ln2b    = (const float*)d_in[15];
    const float* pmW1    = (const float*)d_in[16];
    const float* pmb1    = (const float*)d_in[17];
    const float* pmlng   = (const float*)d_in[18];
    const float* pmlnb   = (const float*)d_in[19];
    const float* pmW2    = (const float*)d_in[20];
    const float* pmb2    = (const float*)d_in[21];
    const float* pmW3    = (const float*)d_in[22];
    const float* pmb3    = (const float*)d_in[23];
    float* out = (float*)d_out;

    float* wsf  = (float*)d_ws;
    float* z0   = wsf;              // TQ*64
    float* z1   = wsf + 131072;     // TQ*64
    float* kb   = wsf + 262144;     // NN*64
    float* vb   = wsf + 327680;     // NN*64
    float* azb  = wsf + 393216;     // TQ*64
    float* aHb  = wsf + 524288;     // NN*64
    float* b2f  = wsf + 589824;     // 64
    __bf16* W2bf = (__bf16*)(wsf + 589888);  // 64*64 bf16 (2048 floats worth)
    float* csumf = wsf + 591936;    // 64

    qe_kernel<<<TQ, 64, 0, stream>>>(qpts, qe_W1, qe_b1, qe_W2, qe_b2, z0);

    float* za = z0;
    float* zb = z1;
    for (int i = 0; i < 2; ++i) {
        const float* Wq   = attn_in  + i * 12288;
        const float* Wk   = attn_in  + i * 12288 + 4096;
        const float* Wv   = attn_in  + i * 12288 + 8192;
        const float* Wout = attn_out + i * 4096;
        kv_kernel<<<NN, 64, 0, stream>>>(H, Wk, Wv, kb, vb);
        attn_layer_kernel<<<TQ, 128, 0, stream>>>(za, zb, kb, vb, Wq, Wout,
                                                  ln1g + i * 64, ln1b + i * 64,
                                                  fW1 + i * 16384, fb1 + i * 256,
                                                  fW2 + i * 16384, fb2 + i * 64,
                                                  ln2g + i * 64, ln2b + i * 64);
        float* tmp = za; za = zb; zb = tmp;
    }
    // za now holds the final z

    pm_prep_kernel<<<TQ + NN + 1, 64, 0, stream>>>(za, H, pmW1, pmb1, pmlng, pmlnb,
                                                   pmW2, pmb2, azb, aHb, W2bf, b2f, csumf);

    pairwise_kernel<<<1024, 256, 0, stream>>>(azb, aHb, W2bf, b2f, csumf, pmW3, pmb3, out);
}
